// COSMO_layer_20521353740838
// MI455X (gfx1250) — matrix operand order repaired and checked by reading, not compiled, not run on hardware
//
#include <hip/hip_runtime.h>
#include <hip/hip_bf16.h>

// ---------------- constants from the reference ----------------
#define NS        51
#define NITERS    50
#define T_K       623.15f
#define Q0_C      79.53f
#define R0_C      66.69f
#define C_HB      85580.0f
#define RGAS_C    (8.3144598f / 4184.0f)
#define SIG_HB    0.0084f
#define EPS_C     3.667f
#define AEFF_C    7.5f
#define EO_C      0.0002395f
#define ZC_C      10.0f
#define X0_C      0.235f
#define X1_C      (1.0f - 0.235f)

typedef __attribute__((ext_vector_type(16))) __bf16 v16bf;
typedef __attribute__((ext_vector_type(2)))  __bf16 v2bf;
typedef __attribute__((ext_vector_type(8)))  float  v8f;
typedef __attribute__((ext_vector_type(8)))  unsigned v8u;

// Slots (t,j) hold row n = 16t + 8*hi + j.  For t==3, j>=3 both halves are
// padding (rows 51-55 / 59-63) -> those slots are dropped everywhere.
__device__ __forceinline__ constexpr int nj(int t) { return t == 3 ? 3 : 8; }

__device__ __forceinline__ unsigned pk_bf16(float a, float b) {
  v2bf t; t[0] = (__bf16)a; t[1] = (__bf16)b;   // -> v_cvt_pk_bf16_f32
  return __builtin_bit_cast(unsigned, t);
}

// =====================================================================
// Kernel 0: build 2*SSS (51x51 padded to 64x64) in WMMA A-fragment lane
// order, bf16.  The factor 2 folds the fixed-point averaging into the
// matrix: rcp(2y) = 0.5*Gnew, so the G update is a single FMA.
// A order used here (contiguous k, same as B): lane L (M=L%16, hi=L/16), element e holds
// K = 32s + 16*hi + e.
// Padding: columns k>=51 -> 0; rows m>=51 -> 2.0 (finite padded fixed point;
// B-fragment pad entries are exact zeros, so no feedback into valid rows).
// Flat bf16 index = ((t*2+s)*32 + lane)*16 + e   (4096 elems, 8KB)
// =====================================================================
__global__ void sss_pack_kernel(__bf16* __restrict__ packed) {
  int base = (blockIdx.x * blockDim.x + threadIdx.x) * 4;
  const float rt    = RGAS_C * T_K;
  const float fpol  = (EPS_C - 1.0f) / (EPS_C + 0.5f);
  const float alpha = 0.3f * (AEFF_C * sqrtf(AEFF_C)) / EO_C;  // 7.5^1.5
  const float ap    = fpol * alpha;
  #pragma unroll
  for (int r = 0; r < 4; ++r) {
    int idx  = base + r;
    if (idx >= 4096) return;
    int e    = idx & 15;
    int lane = (idx >> 4) & 31;
    int frag = idx >> 9;            // 0..7
    int t    = frag >> 1;
    int s    = frag & 1;
    int m    = 16 * t + (lane & 15);
    int hi   = lane >> 4;
    int k    = 32 * s + 16 * hi + e;
    float val;
    if (k >= NS)      val = 0.0f;
    else if (m >= NS) val = 2.0f;
    else {
      float si = -0.025f + 0.001f * (float)m;
      float sj = -0.025f + 0.001f * (float)k;
      float smax = fmaxf(si, sj), smin = fminf(si, sj);
      float dw = 0.5f * ap * (si + sj) * (si + sj)
               + C_HB * fmaxf(0.0f, smax - SIG_HB) * fminf(0.0f, smin + SIG_HB);
      val = 2.0f * __expf(-dw / rt);
    }
    packed[idx] = (__bf16)val;
  }
}

// Half-wave row exchange for the D->B layout conversion.
// v_permlane16_swap_b32: returns {[p0.lo|p1.lo], [p0.hi|p1.hi]}.
__device__ __forceinline__ void half_swap(unsigned p0, unsigned p1, int hi,
                                          unsigned& first, unsigned& second) {
#if __has_builtin(__builtin_amdgcn_permlane16_swap)
  typedef __attribute__((ext_vector_type(2))) unsigned v2u;
  v2u r = __builtin_amdgcn_permlane16_swap(p0, p1, false, false);
  first  = r[0];
  second = r[1];
#else
  unsigned s0 = (unsigned)__shfl_xor((int)p0, 16, 32);
  unsigned s1 = (unsigned)__shfl_xor((int)p1, 16, 32);
  first  = hi ? s1 : p0;
  second = hi ? p1 : s0;
#endif
}

// =====================================================================
// 50-step fixed-point Gamma solve on the WMMA pipe.
// G / psigma in C/D layout; B-fragment 32x16 bf16: lane N = L%16, element e
// holds K = 16*hi + e (+32 per k-step).
// =====================================================================
__device__ __forceinline__ void gamma_solve(const float ps[4][8], float G[4][8],
                                            const v16bf A[4][2], int hi) {
  #pragma unroll
  for (int t = 0; t < 4; ++t)
    #pragma unroll
    for (int j = 0; j < nj(t); ++j)
      G[t][j] = 1.0f;

  #pragma unroll 1
  for (int it = 0; it < NITERS; ++it) {
    // v = psigma .* G packed to bf16 dwords (K-consecutive pairs), lane-local
    unsigned p[4][4];
    #pragma unroll
    for (int t = 0; t < 3; ++t)
      #pragma unroll
      for (int x = 0; x < 4; ++x)
        p[t][x] = pk_bf16(ps[t][2 * x] * G[t][2 * x],
                          ps[t][2 * x + 1] * G[t][2 * x + 1]);
    // tile 3: only j=0..2 live; K in {51..55, 59..63} are exact zeros
    p[3][0] = pk_bf16(ps[3][0] * G[3][0], ps[3][1] * G[3][1]);
    p[3][1] = pk_bf16(ps[3][2] * G[3][2], 0.0f);
    p[3][2] = 0u;
    p[3][3] = 0u;

    v8u d0, d1;
    #pragma unroll
    for (int x = 0; x < 4; ++x) {
      unsigned a, bq;
      half_swap(p[0][x], p[1][x], hi, a, bq);
      d0[x] = a; d0[x + 4] = bq;
      half_swap(p[2][x], p[3][x], hi, a, bq);
      d1[x] = a; d1[x + 4] = bq;
    }
    v16bf B0 = __builtin_bit_cast(v16bf, d0);
    v16bf B1 = __builtin_bit_cast(v16bf, d1);

    #pragma unroll
    for (int t = 0; t < 4; ++t) {
      v8f acc = {};
      acc = __builtin_amdgcn_wmma_f32_16x16x32_bf16(false, A[t][0], false, B0,
                                                    (short)0, acc, false, false);
      acc = __builtin_amdgcn_wmma_f32_16x16x32_bf16(false, A[t][1], false, B1,
                                                    (short)0, acc, false, false);
      #pragma unroll
      for (int j = 0; j < nj(t); ++j) {
        // A holds 2*SSS, so rcp gives 0.5*Gnew -> single-FMA averaging
        float hn = __builtin_amdgcn_rcpf(acc[j]);
        G[t][j]  = __builtin_fmaf(0.5f, G[t][j], hn);
      }
    }
  }
}

// =====================================================================
// Main kernel: one wave per 16 batch columns.
// =====================================================================
__global__ __launch_bounds__(256, 1)
void cosmo_kernel(const float* __restrict__ my_in,   // [B,51]
                  const float* __restrict__ vcomp,   // [B]
                  const float* __restrict__ vt_in,   // [B,51,2]
                  const float* __restrict__ vvt,     // [B]
                  const __bf16* __restrict__ sss,    // packed 2*SSS fragments
                  float* __restrict__ out,           // [B]
                  int Btot) {
  const int lane = threadIdx.x & 31;
  const int hi   = lane >> 4;
  const int wave = (blockIdx.x * blockDim.x + threadIdx.x) >> 5;
  int b = wave * 16 + (lane & 15);
  if (b >= Btot) b = Btot - 1;

  // ---- resident A fragments: 4 m-tiles x 2 k-steps, bf16 (64 VGPRs) ----
  v16bf A[4][2];
  {
    const v16bf* sp = (const v16bf*)sss;
    #pragma unroll
    for (int t = 0; t < 4; ++t)
      #pragma unroll
      for (int s = 0; s < 2; ++s)
        A[t][s] = sp[(t * 2 + s) * 32 + lane];
  }

  // ---- load my / vt in C/D layout, accumulate row sums ----
  float psm[4][8], psp[4][8];     // psigma_mix, psigma_pure
  float A0, A1;
  {
    float myv[4][8], vtv[4][8];
    float pa0 = 0.0f, pa1 = 0.0f;
    #pragma unroll
    for (int t = 0; t < 4; ++t)
      #pragma unroll
      for (int j = 0; j < nj(t); ++j) {
        int n = 16 * t + 8 * hi + j;
        float m = (n < NS) ? my_in[b * NS + n] : 0.0f;
        float v = (n < NS) ? vt_in[(b * NS + n) * 2 + 1] : 0.0f;
        myv[t][j] = m; vtv[t][j] = v;
        pa0 += m; pa1 += v;
      }
    A0 = pa0 + __shfl_xor(pa0, 16, 32);
    A1 = pa1 + __shfl_xor(pa1, 16, 32);
    float invmix = __builtin_amdgcn_rcpf(X0_C * A0 + X1_C * A1);
    float inv0   = __builtin_amdgcn_rcpf(A0);
    #pragma unroll
    for (int t = 0; t < 4; ++t)
      #pragma unroll
      for (int j = 0; j < nj(t); ++j) {
        psm[t][j] = (X0_C * myv[t][j] + X1_C * vtv[t][j]) * invmix;
        psp[t][j] = myv[t][j] * inv0;
      }
  }

  // ---- solve mix; keep lnGamma_mix packed as bf16 pairs ----
  float G[4][8];
  unsigned lmix[4][4];
  gamma_solve(psm, G, A, hi);
  #pragma unroll
  for (int t = 0; t < 3; ++t)
    #pragma unroll
    for (int x = 0; x < 4; ++x)
      lmix[t][x] = pk_bf16(__logf(G[t][2 * x]), __logf(G[t][2 * x + 1]));
  lmix[3][0] = pk_bf16(__logf(G[3][0]), __logf(G[3][1]));
  lmix[3][1] = pk_bf16(__logf(G[3][2]), 0.0f);

  // ---- solve pure, residual sum ----
  gamma_solve(psp, G, A, hi);
  float part = 0.0f;
  #pragma unroll
  for (int t = 0; t < 3; ++t)
    #pragma unroll
    for (int x = 0; x < 4; ++x) {
      v2bf lm = __builtin_bit_cast(v2bf, lmix[t][x]);
      part += psp[t][2 * x]     * ((float)lm[0] - __logf(G[t][2 * x]));
      part += psp[t][2 * x + 1] * ((float)lm[1] - __logf(G[t][2 * x + 1]));
    }
  {
    v2bf lm0 = __builtin_bit_cast(v2bf, lmix[3][0]);
    v2bf lm1 = __builtin_bit_cast(v2bf, lmix[3][1]);
    part += psp[3][0] * ((float)lm0[0] - __logf(G[3][0]));
    part += psp[3][1] * ((float)lm0[1] - __logf(G[3][1]));
    part += psp[3][2] * ((float)lm1[0] - __logf(G[3][2]));
  }
  float sres = part + __shfl_xor(part, 16, 32);
  float lng_resid = (A0 / AEFF_C) * sres;

  // ---- combinatorial term ----
  float q0 = A0 / Q0_C;
  float q1 = A1 / Q0_C;
  float r0 = vcomp[b] / R0_C;
  float r1 = vvt[b] / R0_C;
  float xq = X0_C * q0 + X1_C * q1;
  float xr = X0_C * r0 + X1_C * r1;
  float theta = X0_C * q0 / xq;
  float phi   = X0_C * r0 / xr;
  float l0 = 0.5f * ZC_C * (r0 - q0) - (r0 - 1.0f);
  float l1 = 0.5f * ZC_C * (r1 - q1) - (r1 - 1.0f);
  float xl = X0_C * l0 + X1_C * l1;
  float lng_comb = __logf(phi / X0_C)
                 + 0.5f * ZC_C * q0 * __logf(theta / phi)
                 + l0 - (phi / X0_C) * xl;

  if (!hi) out[b] = lng_resid + lng_comb;
}

// =====================================================================
extern "C" void kernel_launch(void* const* d_in, const int* in_sizes, int n_in,
                              void* d_out, int out_size, void* d_ws, size_t ws_size,
                              hipStream_t stream) {
  const float* my = (const float*)d_in[0];
  const float* vc = (const float*)d_in[1];
  const float* vt = (const float*)d_in[2];
  const float* vv = (const float*)d_in[3];
  int B = in_sizes[1];                    // v_compound is [B]

  __bf16* sss = (__bf16*)d_ws;            // 8 KB packed A fragments
  sss_pack_kernel<<<4, 256, 0, stream>>>(sss);

  int waves   = (B + 15) / 16;
  int threads = waves * 32;
  int block   = 256;
  int grid    = (threads + block - 1) / block;
  cosmo_kernel<<<grid, block, 0, stream>>>(my, vc, vt, vv, sss,
                                           (float*)d_out, B);
}
